// TC_module_38611755991897
// MI455X (gfx1250) — matrix operand order repaired and checked by reading, not compiled, not run on hardware
//
#include <hip/hip_runtime.h>

// ---------------------------------------------------------------------------
// Types
// ---------------------------------------------------------------------------
typedef __bf16 v16bf __attribute__((ext_vector_type(16)));
typedef float  v8f   __attribute__((ext_vector_type(8)));

#define LDSK 40  // padded LDS row stride (bf16 elems): 32 data + 8 pad

__device__ __forceinline__ unsigned short f2bf(float x) {
  unsigned u = __builtin_bit_cast(unsigned, x);
  u += 0x7FFFu + ((u >> 16) & 1u);   // round-to-nearest-even
  return (unsigned short)(u >> 16);
}

union FragU { uint4 u[2]; v16bf v; };

// A fragment (16x32 bf16): lane = g*16 + r; row m=r; k pairs at g*8 and 16+g*8
__device__ __forceinline__ v16bf load_frag_a(const unsigned short* lds, int row0, int lane) {
  const int r = lane & 15, g = lane >> 4;
  const unsigned short* p = lds + (row0 + r) * LDSK + g * 8;
  FragU f;
  f.u[0] = *(const uint4*)(p);
  f.u[1] = *(const uint4*)(p + 16);
  return f.v;
}
// B fragment (32x16 bf16, fed from W^T tile): lane holds n=r, k pairs at g*8 and 16+g*8
__device__ __forceinline__ v16bf load_frag_b(const unsigned short* lds, int row0, int lane) {
  const int r = lane & 15, g = lane >> 4;
  const unsigned short* p = lds + (row0 + r) * LDSK + g * 8;
  FragU f;
  f.u[0] = *(const uint4*)(p);
  f.u[1] = *(const uint4*)(p + 16);
  return f.v;
}

// gfx1250 async global->LDS copy (16B per lane), tracked by ASYNCcnt
__device__ __forceinline__ void async_copy16(unsigned lds_off, const void* gptr) {
  asm volatile("global_load_async_to_lds_b128 %0, %1, off"
               :: "v"(lds_off), "v"((unsigned long long)(size_t)gptr)
               : "memory");
}
__device__ __forceinline__ void wait_async0() {
  asm volatile("s_wait_asynccnt 0x0" ::: "memory");
}

// ---------------------------------------------------------------------------
// Prep kernels
// ---------------------------------------------------------------------------
__global__ void norm_k(const float* __restrict__ W, float* __restrict__ scale) {
  __shared__ float red[256];
  float s = 0.f;
  for (int i = threadIdx.x; i < 512 * 256; i += 256) { float v = W[i]; s += v * v; }
  red[threadIdx.x] = s;
  __syncthreads();
  for (int o = 128; o > 0; o >>= 1) {
    if ((int)threadIdx.x < o) red[threadIdx.x] += red[threadIdx.x + o];
    __syncthreads();
  }
  if (threadIdx.x == 0) *scale = 1.0f / fmaxf(sqrtf(red[0]), 1.0f);
}

// out[n*ldo + koff + k] = bf16(in[k*N + n] * s)   (weight transpose+convert)
__global__ void tr_cvt(const float* __restrict__ in, unsigned short* __restrict__ out,
                       int K, int N, int ldo, int koff, const float* __restrict__ sp) {
  const float s = sp ? *sp : 1.0f;
  const int total = K * N;
  const int stride = gridDim.x * blockDim.x;
  for (int i = blockIdx.x * blockDim.x + threadIdx.x; i < total; i += stride) {
    int k = i / N, n = i - k * N;
    out[(size_t)n * ldo + koff + k] = f2bf(in[i] * s);
  }
}

// activations -> bf16 staging buffers
__global__ void cvt_act(const float* __restrict__ inp, const float* __restrict__ cen,
                        const float* __restrict__ st, unsigned short* __restrict__ MI,
                        unsigned short* __restrict__ CB, unsigned short* __restrict__ AB) {
  const int stride = gridDim.x * blockDim.x;
  const int n1 = 8192 * 512;
  for (int i = blockIdx.x * blockDim.x + threadIdx.x; i < n1; i += stride) {
    int b = i >> 9, j = i & 511;
    MI[(size_t)b * 768 + j] = f2bf(inp[i]);   // inputs into concat slot [0,512)
    CB[i] = f2bf(cen[i]);                     // center_state
  }
  const int n2 = 8192 * 1024;
  for (int i = blockIdx.x * blockDim.x + threadIdx.x; i < n2; i += stride) {
    int b = i >> 10, j = i & 1023;
    AB[(size_t)b * 2048 + 1024 + j] = f2bf(st[i]);  // module_state into [rnn_in|state]
  }
}

// ---------------------------------------------------------------------------
// bf16 WMMA GEMM: C(MxN) = epilogue(A(MxK,row) @ Wt(NxK,row)^T + bias)
// 128x128 block tile, BK=32, 256 threads (8 waves, wave tile 32x64 = 2x4 frags)
// Async double-buffered LDS: global_load_async_to_lds_b128 streams the next
// tile while WMMAs consume the current one; drained with s_wait_asynccnt 0.
// MODE 0: -> bf16            MODE 1: relu -> bf16
// MODE 2: tanh -> bf16 + f32 MODE 3: relu -> f32 split at n==512
// ---------------------------------------------------------------------------
template<int MODE>
__global__ __launch_bounds__(256) void gemm_bf16(
    const unsigned short* __restrict__ A, int lda,
    const unsigned short* __restrict__ Wt, int K,
    const float* __restrict__ bias,
    unsigned short* __restrict__ outb, int ldob, int ocol,
    float* __restrict__ outf0, float* __restrict__ outf1)
{
  __shared__ __align__(16) unsigned short ldsA[2][128 * LDSK];
  __shared__ __align__(16) unsigned short ldsB[2][128 * LDSK];

  const int tid  = threadIdx.x;
  const int lane = tid & 31;
  const int wv   = tid >> 5;
  const int wy   = wv >> 1;  // 0..3 (rows)
  const int wx   = wv & 1;   // 0..1 (cols)
  const long blockM = (long)blockIdx.y * 128;
  const int  blockN = blockIdx.x * 128;

  const int r0 = tid >> 2;            // tile row for chunk i=0 (0..63)
  const int c8 = (tid & 3) * 8;       // k sub-offset (8 bf16 = 16B)

  // per-thread global source pointers (advance by 64B per K-chunk)
  const char* gA0 = (const char*)(A  + blockM * lda        + (long)r0        * lda + c8);
  const char* gA1 = (const char*)(A  + blockM * lda        + (long)(r0 + 64) * lda + c8);
  const char* gB0 = (const char*)(Wt + (long)blockN * K    + (long)r0        * K   + c8);
  const char* gB1 = (const char*)(Wt + (long)blockN * K    + (long)(r0 + 64) * K   + c8);

  // per-thread LDS destinations for both buffers (32-bit LDS byte addresses)
  unsigned lA0[2], lA1[2], lB0[2], lB1[2];
  #pragma unroll
  for (int b = 0; b < 2; ++b) {
    lA0[b] = (unsigned)(size_t)&ldsA[b][r0 * LDSK + c8];
    lA1[b] = (unsigned)(size_t)&ldsA[b][(r0 + 64) * LDSK + c8];
    lB0[b] = (unsigned)(size_t)&ldsB[b][r0 * LDSK + c8];
    lB1[b] = (unsigned)(size_t)&ldsB[b][(r0 + 64) * LDSK + c8];
  }

  v8f acc[2][4];
  #pragma unroll
  for (int mi = 0; mi < 2; ++mi)
    #pragma unroll
    for (int ni = 0; ni < 4; ++ni)
      acc[mi][ni] = {};

  const int KI = K >> 5;

  // prologue: async-fill buffer 0 with tile 0, drain, barrier
  async_copy16(lA0[0], gA0);
  async_copy16(lA1[0], gA1);
  async_copy16(lB0[0], gB0);
  async_copy16(lB1[0], gB1);
  wait_async0();
  __syncthreads();

  for (int kc = 0; kc < KI; ++kc) {
    const int buf = kc & 1;
    if (kc + 1 < KI) {  // stream next tile into the other buffer
      const size_t boff = (size_t)(kc + 1) * 64;  // 32 bf16 = 64 bytes
      async_copy16(lA0[buf ^ 1], gA0 + boff);
      async_copy16(lA1[buf ^ 1], gA1 + boff);
      async_copy16(lB0[buf ^ 1], gB0 + boff);
      async_copy16(lB1[buf ^ 1], gB1 + boff);
    }
    v16bf af[2], bfr[4];
    #pragma unroll
    for (int mi = 0; mi < 2; ++mi)
      af[mi] = load_frag_a(ldsA[buf], wy * 32 + mi * 16, lane);
    #pragma unroll
    for (int ni = 0; ni < 4; ++ni)
      bfr[ni] = load_frag_b(ldsB[buf], wx * 64 + ni * 16, lane);
    #pragma unroll
    for (int mi = 0; mi < 2; ++mi)
      #pragma unroll
      for (int ni = 0; ni < 4; ++ni)
        acc[mi][ni] = __builtin_amdgcn_wmma_f32_16x16x32_bf16(
            false, af[mi], false, bfr[ni], (short)0, acc[mi][ni], false, false);
    wait_async0();     // next tile fully in LDS (per-wave), then block barrier
    __syncthreads();
  }

  // epilogue — C/D layout: n = lane%16, m = v + 8*(lane/16)
  const int nl = lane & 15;
  const int gband = (lane >> 4) * 8;
  #pragma unroll
  for (int ni = 0; ni < 4; ++ni) {
    const int gn = blockN + wx * 64 + ni * 16 + nl;
    const float bv = (MODE == 0) ? 0.0f : bias[gn];
    #pragma unroll
    for (int mi = 0; mi < 2; ++mi) {
      #pragma unroll
      for (int v = 0; v < 8; ++v) {
        const long gm = blockM + wy * 32 + mi * 16 + v + gband;
        float x = acc[mi][ni][v] + bv;
        if (MODE == 1 || MODE == 3) x = fmaxf(x, 0.0f);
        if (MODE == 2) x = tanhf(x);
        if (MODE == 0 || MODE == 1 || MODE == 2)
          outb[gm * ldob + ocol + gn] = f2bf(x);
        if (MODE == 2)
          outf0[gm * 1024 + gn] = x;                 // h_new (fp32)
        if (MODE == 3) {
          if (gn < 512) outf0[gm * 512 + gn] = x;    // network_output
          else          outf1[gm * 256 + (gn - 512)] = x;  // to_center
        }
      }
    }
  }
}

// ---------------------------------------------------------------------------
// Launch
// ---------------------------------------------------------------------------
extern "C" void kernel_launch(void* const* d_in, const int* in_sizes, int n_in,
                              void* d_out, int out_size, void* d_ws, size_t ws_size,
                              hipStream_t stream) {
  (void)in_sizes; (void)n_in; (void)out_size; (void)ws_size;
  const float* inputs = (const float*)d_in[0];
  const float* center = (const float*)d_in[1];
  const float* state  = (const float*)d_in[2];
  const float* W_read = (const float*)d_in[3];
  const float* W_pre  = (const float*)d_in[4];
  const float* b_pre  = (const float*)d_in[5];
  const float* Wx     = (const float*)d_in[6];
  const float* Wh     = (const float*)d_in[7];
  const float* b_rnn  = (const float*)d_in[8];
  const float* W_post = (const float*)d_in[9];
  const float* b_post = (const float*)d_in[10];
  float* out = (float*)d_out;

  char* ws = (char*)d_ws;
  size_t off = 0;
  auto alloc = [&](size_t bytes) -> char* {
    char* p = ws + off; off += (bytes + 255) & ~(size_t)255; return p;
  };
  float*          scale   = (float*)alloc(4);
  unsigned short* Wt_clip = (unsigned short*)alloc((size_t)256 * 512 * 2);   // N x K
  unsigned short* Wt_pre  = (unsigned short*)alloc((size_t)1024 * 768 * 2);
  unsigned short* Wt_xh   = (unsigned short*)alloc((size_t)1024 * 2048 * 2); // [Wx;Wh]^T
  unsigned short* Wt_post = (unsigned short*)alloc((size_t)768 * 1024 * 2);
  unsigned short* CB      = (unsigned short*)alloc((size_t)8192 * 512 * 2);  // center bf16
  unsigned short* MI      = (unsigned short*)alloc((size_t)8192 * 768 * 2);  // [inputs|context]
  unsigned short* AB      = (unsigned short*)alloc((size_t)8192 * 2048 * 2); // [rnn_in|state]
  unsigned short* Hb      = (unsigned short*)alloc((size_t)8192 * 1024 * 2); // h_new bf16

  norm_k<<<1, 256, 0, stream>>>(W_read, scale);
  tr_cvt<<<256,  256, 0, stream>>>(W_read, Wt_clip, 512, 256, 512, 0, scale);
  tr_cvt<<<1024, 256, 0, stream>>>(W_pre,  Wt_pre,  768, 1024, 768, 0, nullptr);
  tr_cvt<<<2048, 256, 0, stream>>>(Wx,     Wt_xh,  1024, 1024, 2048, 0, nullptr);
  tr_cvt<<<2048, 256, 0, stream>>>(Wh,     Wt_xh,  1024, 1024, 2048, 1024, nullptr);
  tr_cvt<<<1024, 256, 0, stream>>>(W_post, Wt_post, 1024, 768, 1024, 0, nullptr);
  cvt_act<<<2048, 256, 0, stream>>>(inputs, center, state, MI, CB, AB);

  // context = center @ W_clip  -> MI[:, 512:768]
  gemm_bf16<0><<<dim3(2, 64), 256, 0, stream>>>(CB, 512, Wt_clip, 512, nullptr,
                                                MI, 768, 512, nullptr, nullptr);
  // rnn_in = relu(MI @ W_pre + b_pre) -> AB[:, 0:1024]
  gemm_bf16<1><<<dim3(8, 64), 256, 0, stream>>>(MI, 768, Wt_pre, 768, b_pre,
                                                AB, 2048, 0, nullptr, nullptr);
  // h = tanh([rnn_in|state] @ [Wx;Wh] + b_rnn) -> Hb (bf16) + d_out h region (f32)
  gemm_bf16<2><<<dim3(8, 64), 256, 0, stream>>>(AB, 2048, Wt_xh, 2048, b_rnn,
                                                Hb, 1024, 0,
                                                out + (size_t)8192 * 768, nullptr);
  // module_output = relu(h @ W_post + b_post) -> net / o2c split (f32)
  gemm_bf16<3><<<dim3(6, 64), 256, 0, stream>>>(Hb, 1024, Wt_post, 1024, b_post,
                                                nullptr, 0, 0,
                                                out, out + (size_t)8192 * 512);
}
